// RCEncoding_26645977104957
// MI455X (gfx1250) — matrix operand order repaired and checked by reading, not compiled, not run on hardware
//
#include <hip/hip_runtime.h>
#include <hip/hip_bf16.h>

// Problem constants (match reference)
#define RN    65536
#define RDIM  512
#define RM    4
#define RK    256

typedef __attribute__((ext_vector_type(16))) __bf16 v16bf;
typedef __attribute__((ext_vector_type(8)))  float  v8f;

// ---------- helpers ----------
__device__ __forceinline__ unsigned short f2bf(float f) {
    unsigned int u = __builtin_bit_cast(unsigned int, f);
    unsigned int r = (u + 0x7FFFu + ((u >> 16) & 1u)) >> 16;  // RNE
    return (unsigned short)r;
}

__device__ __forceinline__ float bf2f(unsigned short h) {
    unsigned int u = ((unsigned int)h) << 16;
    return __builtin_bit_cast(float, u);
}

__device__ __forceinline__ float gumbel_noise(int n, int k, int stage) {
    unsigned int h = (unsigned int)n * 0x9E3779B1u
                   ^ (unsigned int)k * 0x85EBCA77u
                   ^ (unsigned int)(stage + 1) * 0xC2B2AE3Du;
    h ^= h >> 16; h *= 0x7FEB352Du;
    h ^= h >> 15; h *= 0x846CA68Bu;
    h ^= h >> 16;
    float u = (float)(h >> 8) * 5.9604645e-08f + 1e-9f;  // (0,1)
    return -__logf(-__logf(u));
}

// ---------- precompute kernels (tiny: ~0.8 GFLOP total) ----------

// c_proj[i][k][d] = sum_e codebooks[i][k][e] * W_att[i][e][d]     [4,256,512]
__global__ void k_cproj(const float* __restrict__ codebooks,
                        const float* __restrict__ W_att,
                        float* __restrict__ cproj) {
    int t = blockIdx.x * 256 + threadIdx.x;
    if (t >= RM * RK * RDIM) return;
    int i = t >> 17, rem = t & 131071, k = rem >> 9, d = rem & 511;
    const float* crow = codebooks + ((size_t)i * RK + k) * RDIM;
    const float* w    = W_att + (size_t)i * RDIM * RDIM + d;
    float s = 0.f;
    for (int e = 0; e < RDIM; ++e) s += crow[e] * w[(size_t)e * RDIM];
    cproj[t] = s;
}

// B_i[dd][nn] = sum_e W_att[i][dd][e] * cproj[i][nn][e], stored directly in
// WMMA-B (32x16 bf16, wave32) fragment order:
//   frag (i,kc,nt): 32 lanes x 16 bf16; lane<16 -> n=nt*16+lane, k=kc*32+0..15
//                                       lane>=16 -> n=nt*16+lane-16, k=kc*32+16..31
//   within lane, ushort position == (k & 15)   (vgpr j holds k=2j,2j+1)
__global__ void k_bpack(const float* __restrict__ W_att,
                        const float* __restrict__ cproj,
                        unsigned short* __restrict__ Bpack) {
    int t = blockIdx.x * 256 + threadIdx.x;
    if (t >= RM * RDIM * RK) return;
    int i = t >> 17, rem = t & 131071, dd = rem >> 8, nn = rem & 255;
    const float* wrow = W_att + ((size_t)i * RDIM + dd) * RDIM;
    const float* crow = cproj + ((size_t)i * RK + nn) * RDIM;
    float s = 0.f;
    for (int e = 0; e < RDIM; ++e) s += wrow[e] * crow[e];
    int kc = dd >> 5, kin = dd & 31, nt = nn >> 4, nl = nn & 15;
    int lanep = (kin < 16) ? nl : (nl + 16);
    int kk = kin & 15;
    size_t off = ((((size_t)(i * 16 + kc) * 16 + nt) * 32 + lanep) * 16) + kk;
    Bpack[off] = f2bf(s);
}

// CR[i][k][d] = sum_e codebooks[i][k][e] * W_rnn[e][d]            [4,256,512] f32
__global__ void k_cr(const float* __restrict__ codebooks,
                     const float* __restrict__ W_rnn,
                     float* __restrict__ CR) {
    int t = blockIdx.x * 256 + threadIdx.x;
    if (t >= RM * RK * RDIM) return;
    int i = t >> 17, rem = t & 131071, k = rem >> 9, d = rem & 511;
    const float* crow = codebooks + ((size_t)i * RK + k) * RDIM;
    float s = 0.f;
    for (int e = 0; e < RDIM; ++e) s += crow[e] * W_rnn[(size_t)e * RDIM + d];
    CR[t] = s;
}

// ---------- fused main kernel ----------
// 32 rows / block, 8 wave32s. Wave w handles BOTH 16-row M-blocks and the
// 32-column slab [w*32, w*32+32): each B fragment fetched once per WG and
// reused by 2 WMMAs -> B_i is read from L2 exactly once per workgroup.
// x tile lives ONLY as the bf16 LDS mirror (16B XOR swizzle); the residual
// update is an LDS bf16 RMW. Keeps peak VGPRs < 256 for occupancy.
#define TROWS 32

__global__ __launch_bounds__(256) void rc_main(const float* __restrict__ x,
                                               const float* __restrict__ codebooks,
                                               const unsigned short* __restrict__ Bpack,
                                               const float* __restrict__ CR,
                                               float* __restrict__ out) {
    __shared__ unsigned short xs[TROWS * RDIM];  // 32 KB, swizzled bf16
    __shared__ float red_nv[TROWS][8];
    __shared__ int   red_ni[TROWS][8];
    __shared__ float red_cv[TROWS][8];
    __shared__ int   red_ci[TROWS][8];
    __shared__ int   sel_all[RM][TROWS];

    const int tid  = threadIdx.x;
    const int lane = tid & 31;
    const int wave = tid >> 5;
    const int wg_base = blockIdx.x * TROWS;

    // ---- load x tile (coalesced, non-temporal: keep L2 for B/CR/codebooks),
    //      convert to bf16, store swizzled into LDS ----
#pragma unroll 4
    for (int j = 0; j < 64; ++j) {
        int flat = tid + 256 * j;
        float v = __builtin_nontemporal_load(&x[(size_t)wg_base * RDIM + flat]);
        int row = flat >> 9, col = flat & 511;
        int c = (col >> 3) ^ (row & 15);
        xs[row * RDIM + (c << 3) + (col & 7)] = f2bf(v);
    }
    __syncthreads();

    const int kbl = (lane < 16) ? 0 : 8;  // A-layout lane K base

    for (int i = 0; i < RM; ++i) {
        // ---- GEMM: logits(32 rows x 32 cols per wave) = x_tile @ B_i ----
        v8f acc[2][2] = {};
#pragma unroll 2
        for (int kc = 0; kc < 16; ++kc) {
            union { unsigned int u[8]; v16bf v; } af[2];
#pragma unroll
            for (int mb = 0; mb < 2; ++mb) {
                int arow = mb * 16 + (lane & 15);
#pragma unroll
                for (int j = 0; j < 8; ++j) {
                    int k = kc * 32 + 2 * kbl + 2 * j;
                    int c = (k >> 3) ^ (arow & 15);
                    af[mb].u[j] = *(const unsigned int*)&xs[arow * RDIM + (c << 3) + (k & 7)];
                }
            }
            union { uint4 u4[2]; v16bf v; } bfr[2];
#pragma unroll
            for (int nt = 0; nt < 2; ++nt) {
                const uint4* bp = (const uint4*)(Bpack +
                    ((((size_t)(i * 16 + kc) * 16 + (wave * 2 + nt)) * 32 + lane) * 16));
                bfr[nt].u4[0] = bp[0];
                bfr[nt].u4[1] = bp[1];
            }
#pragma unroll
            for (int mb = 0; mb < 2; ++mb)
#pragma unroll
                for (int nt = 0; nt < 2; ++nt)
                    acc[mb][nt] = __builtin_amdgcn_wmma_f32_16x16x32_bf16(
                        false, af[mb].v, false, bfr[nt].v, (short)0, acc[mb][nt], false, false);
        }

        // ---- noisy argmax (selection) + clean argmax (indexes output) ----
        float nv[2][8], cv[2][8]; int ni[2][8], ci[2][8];
#pragma unroll
        for (int mb = 0; mb < 2; ++mb)
#pragma unroll
            for (int r = 0; r < 8; ++r) {
                nv[mb][r] = -1e30f; cv[mb][r] = -1e30f; ni[mb][r] = 0; ci[mb][r] = 0;
            }
#pragma unroll
        for (int nt = 0; nt < 2; ++nt) {
            int col = wave * 32 + nt * 16 + (lane & 15);
#pragma unroll
            for (int mb = 0; mb < 2; ++mb)
#pragma unroll
                for (int r = 0; r < 8; ++r) {
                    int rowm = mb * 16 + r + ((lane < 16) ? 0 : 8);   // C-layout row
                    int n = wg_base + rowm;
                    float logit = acc[mb][nt][r];
                    float noisy = logit + gumbel_noise(n, col, i);
                    if (noisy > nv[mb][r]) { nv[mb][r] = noisy; ni[mb][r] = col; }
                    if (logit > cv[mb][r]) { cv[mb][r] = logit; ci[mb][r] = col; }
                }
        }
#pragma unroll
        for (int m = 1; m < 16; m <<= 1) {
#pragma unroll
            for (int mb = 0; mb < 2; ++mb)
#pragma unroll
                for (int r = 0; r < 8; ++r) {
                    float onv = __shfl_xor(nv[mb][r], m, 16); int oni = __shfl_xor(ni[mb][r], m, 16);
                    float ocv = __shfl_xor(cv[mb][r], m, 16); int oci = __shfl_xor(ci[mb][r], m, 16);
                    if (onv > nv[mb][r] || (onv == nv[mb][r] && oni < ni[mb][r])) { nv[mb][r] = onv; ni[mb][r] = oni; }
                    if (ocv > cv[mb][r] || (ocv == cv[mb][r] && oci < ci[mb][r])) { cv[mb][r] = ocv; ci[mb][r] = oci; }
                }
        }
        if ((lane & 15) == 0) {
#pragma unroll
            for (int mb = 0; mb < 2; ++mb) {
                int base_m = mb * 16 + ((lane == 0) ? 0 : 8);
#pragma unroll
                for (int r = 0; r < 8; ++r) {
                    red_nv[base_m + r][wave] = nv[mb][r]; red_ni[base_m + r][wave] = ni[mb][r];
                    red_cv[base_m + r][wave] = cv[mb][r]; red_ci[base_m + r][wave] = ci[mb][r];
                }
            }
        }
        __syncthreads();
        if (tid < TROWS) {
            float bn = red_nv[tid][0]; int bni = red_ni[tid][0];
            float bc = red_cv[tid][0]; int bci = red_ci[tid][0];
#pragma unroll
            for (int w = 1; w < 8; ++w) {
                float vn = red_nv[tid][w]; int in_ = red_ni[tid][w];
                float vc = red_cv[tid][w]; int ic_ = red_ci[tid][w];
                if (vn > bn || (vn == bn && in_ < bni)) { bn = vn; bni = in_; }
                if (vc > bc || (vc == bc && ic_ < bci)) { bc = vc; bci = ic_; }
            }
            sel_all[i][tid] = bni;
            // indexes output: int32 region after the [N,DIM] f32 block; layout [N,M]
            ((int*)out)[(size_t)RN * RDIM + (size_t)(wg_base + tid) * RM + i] = bci;
        }
        __syncthreads();

        // ---- residual update: xs <- xs - CR_i[sel]  (LDS bf16 RMW; CR L2-resident) ----
        if (i < RM - 1) {
#pragma unroll 4
            for (int j = 0; j < 64; ++j) {
                int flat = tid + 256 * j;
                int row = flat >> 9, col = flat & 511;
                int s = sel_all[i][row];
                int c = (col >> 3) ^ (row & 15);
                unsigned short* p = &xs[row * RDIM + (c << 3) + (col & 7)];
                float v = bf2f(*p) - CR[((size_t)(i * RK + s)) * RDIM + col];
                *p = f2bf(v);
            }
            __syncthreads();
        }
    }

    // ---- out_sum = sum_i codebooks[i][sel_i]; non-temporal store (streamed once) ----
#pragma unroll 4
    for (int j = 0; j < 64; ++j) {
        int flat = tid + 256 * j;
        int row = flat >> 9, col = flat & 511;
        float s = 0.f;
#pragma unroll
        for (int i2 = 0; i2 < RM; ++i2) {
            s += codebooks[((size_t)(i2 * RK + sel_all[i2][row])) * RDIM + col];
        }
        __builtin_nontemporal_store(s, &out[(size_t)wg_base * RDIM + flat]);
    }
}

// ---------- launcher ----------
extern "C" void kernel_launch(void* const* d_in, const int* in_sizes, int n_in,
                              void* d_out, int out_size, void* d_ws, size_t ws_size,
                              hipStream_t stream) {
    const float* x         = (const float*)d_in[0];
    const float* codebooks = (const float*)d_in[1];
    const float* W_att     = (const float*)d_in[2];
    const float* W_rnn     = (const float*)d_in[3];
    float* out = (float*)d_out;

    char* ws = (char*)d_ws;
    float*          cproj = (float*)ws;                                      // 2 MB
    unsigned short* Bpack = (unsigned short*)(ws + (size_t)2 * 1024 * 1024); // 1 MB
    float*          CR    = (float*)(ws + (size_t)3 * 1024 * 1024);          // 2 MB

    const int tp = RM * RK * RDIM;  // 524288
    k_cproj<<<(tp + 255) / 256, 256, 0, stream>>>(codebooks, W_att, cproj);
    k_bpack<<<(tp + 255) / 256, 256, 0, stream>>>(W_att, cproj, Bpack);
    k_cr   <<<(tp + 255) / 256, 256, 0, stream>>>(codebooks, W_rnn, CR);

    rc_main<<<RN / TROWS, 256, 0, stream>>>(x, codebooks, Bpack, CR, out);
}
